// StabilityAugmentedMemory_29935922053583
// MI455X (gfx1250) — matrix operand order repaired and checked by reading, not compiled, not run on hardware
//
#include <hip/hip_runtime.h>
#include <hip/hip_bf16.h>
#include <math.h>

typedef __bf16 bf16_t;
typedef __bf16 v16bf __attribute__((ext_vector_type(16)));
typedef __bf16 v8bf  __attribute__((ext_vector_type(8)));
typedef float  v8f   __attribute__((ext_vector_type(8)));

#define DD   128
#define EE   64
#define TT   64
#define PP   5
#define KC   256          // folded qi width: raw(128) + ef(64) + te(64)
#define ROWS 32           // batch rows per block
#define LNE  1e-4f

// ---------------------------------------------------------------------------
// Kernel 0: fold edge projection into query weights, emit bf16 hi/lo splits.
//   Wc[n][0:128)   = query_w[n][0:128)                      (raw part)
//   Wc[n][128:192) = sum_d query_w[n][128+d] * edge_w[d][j] (edge folded)
//   Wc[n][192:256) = query_w[n][256+j]                      (time part)
//   bcomb[n]       = query_b[n] + sum_d query_w[n][128+d] * edge_b[d]
// ---------------------------------------------------------------------------
__global__ __launch_bounds__(256)
void fold_weights_kernel(const float* __restrict__ qw, const float* __restrict__ qb,
                         const float* __restrict__ ew, const float* __restrict__ eb,
                         bf16_t* __restrict__ Wh, bf16_t* __restrict__ Wl,
                         float* __restrict__ bcomb) {
  const int n = blockIdx.x;    // 0..127
  const int k = threadIdx.x;   // 0..255
  float w;
  if (k < 128) {
    w = qw[n * 320 + k];
  } else if (k < 192) {
    const int j = k - 128;
    float s = 0.f;
    for (int d = 0; d < 128; ++d) s += qw[n * 320 + 128 + d] * ew[d * 64 + j];
    w = s;
  } else {
    w = qw[n * 320 + 64 + k];  // 256 + (k-192)
  }
  const bf16_t hi = (bf16_t)w;
  Wh[n * KC + k] = hi;
  Wl[n * KC + k] = (bf16_t)(w - (float)hi);
  if (k == 0) {
    float s = qb[n];
    for (int d = 0; d < 128; ++d) s += qw[n * 320 + 128 + d] * eb[d];
    bcomb[n] = s;
  }
}

// ---------------------------------------------------------------------------
// wave32 butterfly reductions
// ---------------------------------------------------------------------------
__device__ __forceinline__ float wred(float x) {
  #pragma unroll
  for (int o = 16; o > 0; o >>= 1) x += __shfl_xor(x, o, 32);
  return x;
}
__device__ __forceinline__ void wred2(float& a, float& b) {
  #pragma unroll
  for (int o = 16; o > 0; o >>= 1) {
    a += __shfl_xor(a, o, 32);
    b += __shfl_xor(b, o, 32);
  }
}
__device__ __forceinline__ float clip(float x, float lo, float hi) {
  return fminf(fmaxf(x, lo), hi);
}

// ---------------------------------------------------------------------------
// Kernel 1: fused memory update. 32 rows / block, 256 threads (8 waves).
// ---------------------------------------------------------------------------
__global__ __launch_bounds__(256)
void fused_memory_update(const int*   __restrict__ node_ids,
                         const float* __restrict__ ef,
                         const float* __restrict__ tvals,
                         const float* __restrict__ raw_memory,
                         const float* __restrict__ protos,
                         const float* __restrict__ pg, const float* __restrict__ pbv,
                         const float* __restrict__ time_w, const float* __restrict__ time_b,
                         const bf16_t* __restrict__ Wh, const bf16_t* __restrict__ Wl,
                         const float* __restrict__ bcomb,
                         const float* __restrict__ cg, const float* __restrict__ cbv,
                         const float* __restrict__ gate_w, const float* __restrict__ gate_b,
                         const float* __restrict__ temperature,
                         float* __restrict__ out) {
  __shared__ bf16_t sAh[ROWS][KC + 8];
  __shared__ bf16_t sAl[ROWS][KC + 8];
  __shared__ float  sRaw[ROWS][DD];
  __shared__ float  sTe[ROWS][TT];
  __shared__ float  sQ[ROWS][DD + 4];
  __shared__ int    sNid[ROWS];

  const int tid  = threadIdx.x;
  const int lane = tid & 31;
  const int wave = tid >> 5;
  const int r0   = blockIdx.x * ROWS;

  if (tid < ROWS) sNid[tid] = node_ids[r0 + tid];
  __syncthreads();

  // ------------------- prologue: build qi tiles in LDS --------------------
  for (int it = 0; it < 4; ++it) {
    const int rr  = wave + it * 8;
    const int gr  = r0 + rr;
    const int nid = sNid[rr];
    const float* rp = raw_memory + (long long)nid * DD;
    #pragma unroll
    for (int i = 0; i < 4; ++i) {
      const int c = lane + 32 * i;
      const float x = rp[c];
      sRaw[rr][c] = x;
      const bf16_t h = (bf16_t)x;
      sAh[rr][c] = h;
      sAl[rr][c] = (bf16_t)(x - (float)h);
    }
    #pragma unroll
    for (int i = 0; i < 2; ++i) {
      const int j = lane + 32 * i;
      const float x = ef[(long long)gr * EE + j];
      const bf16_t h = (bf16_t)x;
      sAh[rr][DD + j] = h;
      sAl[rr][DD + j] = (bf16_t)(x - (float)h);
    }
    const float tv = tvals[gr];
    #pragma unroll
    for (int i = 0; i < 2; ++i) {
      const int j = lane + 32 * i;
      const float x = cosf(tv * time_w[j] + time_b[j]);
      sTe[rr][j] = x;
      const bf16_t h = (bf16_t)x;
      sAh[rr][DD + EE + j] = h;
      sAl[rr][DD + EE + j] = (bf16_t)(x - (float)h);
    }
    // prefetch the 20 prototype cachelines for this row (overlap with GEMM)
    if (lane < 20) {
      const float* pp = protos + (long long)nid * (PP * DD) + (lane >> 2) * DD + (lane & 3) * 32;
      __builtin_prefetch(pp, 0, 3);
    }
  }
  __syncthreads();

  // ------------------- WMMA GEMM: [32 x 256] @ [256 x 128] ----------------
  const int half = lane >> 4;
  const int l16  = lane & 15;
  const int rt   = wave >> 2;        // row tile 0/1
  const int nb0  = (wave & 3) * 2;   // two n-blocks of 16 per wave
  const int arow = rt * 16 + l16;

  v8f acc0 = {};
  v8f acc1 = {};

  #pragma unroll
  for (int ks = 0; ks < 8; ++ks) {
    // A fragment: lane holds row (l16), K runs [K0+16*half, K0+16*half+16)
    union { v16bf v; v8bf h[2]; } Ah, Al;
    const bf16_t* ap  = &sAh[arow][ks * 32 + 16 * half];
    const bf16_t* alp = &sAl[arow][ks * 32 + 16 * half];
    Ah.h[0] = *(const v8bf*)(ap);
    Ah.h[1] = *(const v8bf*)(ap + 8);
    Al.h[0] = *(const v8bf*)(alp);
    Al.h[1] = *(const v8bf*)(alp + 8);

    {
      const int n = (nb0 + 0) * 16 + l16;
      const v16bf Bh = *(const v16bf*)(Wh + n * KC + ks * 32 + 16 * half);
      const v16bf Bl = *(const v16bf*)(Wl + n * KC + ks * 32 + 16 * half);
      acc0 = __builtin_amdgcn_wmma_f32_16x16x32_bf16(false, Ah.v, false, Bh, (short)0, acc0, false, false);
      acc0 = __builtin_amdgcn_wmma_f32_16x16x32_bf16(false, Ah.v, false, Bl, (short)0, acc0, false, false);
      acc0 = __builtin_amdgcn_wmma_f32_16x16x32_bf16(false, Al.v, false, Bh, (short)0, acc0, false, false);
    }
    {
      const int n = (nb0 + 1) * 16 + l16;
      const v16bf Bh = *(const v16bf*)(Wh + n * KC + ks * 32 + 16 * half);
      const v16bf Bl = *(const v16bf*)(Wl + n * KC + ks * 32 + 16 * half);
      acc1 = __builtin_amdgcn_wmma_f32_16x16x32_bf16(false, Ah.v, false, Bh, (short)0, acc1, false, false);
      acc1 = __builtin_amdgcn_wmma_f32_16x16x32_bf16(false, Ah.v, false, Bl, (short)0, acc1, false, false);
      acc1 = __builtin_amdgcn_wmma_f32_16x16x32_bf16(false, Al.v, false, Bh, (short)0, acc1, false, false);
    }
  }

  #pragma unroll
  for (int v = 0; v < 8; ++v) {
    const int row = rt * 16 + 8 * half + v;
    sQ[row][(nb0 + 0) * 16 + l16] = acc0[v];
    sQ[row][(nb0 + 1) * 16 + l16] = acc1[v];
  }
  __syncthreads();

  // ------------------- post: LN/tanh, proto attention, gate ---------------
  const float temp = clip(temperature[0], 0.05f, 2.0f) + 1e-4f;

  for (int it = 0; it < 4; ++it) {
    const int rr  = wave + it * 8;
    const int gr  = r0 + rr;
    const int nid = sNid[rr];

    int   c[4];
    float qp[4];
    #pragma unroll
    for (int i = 0; i < 4; ++i) {
      c[i]  = lane + 32 * i;
      qp[i] = sQ[rr][c[i]] + bcomb[c[i]];
    }
    // query = tanh(LN(qp))
    float s1 = qp[0] + qp[1] + qp[2] + qp[3];
    float s2 = qp[0]*qp[0] + qp[1]*qp[1] + qp[2]*qp[2] + qp[3]*qp[3];
    wred2(s1, s2);
    const float mu  = s1 * (1.f / 128.f);
    const float var = s2 * (1.f / 128.f) - mu * mu;
    const float inv = rsqrtf(var + LNE);
    float q[4], qq = 0.f;
    #pragma unroll
    for (int i = 0; i < 4; ++i) {
      q[i] = tanhf((qp[i] - mu) * inv * cg[c[i]] + cbv[c[i]]);
      qq  += q[i] * q[i];
    }
    qq = wred(qq);
    const float qscale = 1.f / fmaxf(sqrtf(qq), 1e-6f);

    // prototypes: LN, cosine sim
    const float* pbase = protos + (long long)nid * (PP * DD);
    float pn[PP][4];
    float sim[PP];
    #pragma unroll
    for (int p = 0; p < PP; ++p) {
      float x[4];
      float a = 0.f, b = 0.f;
      #pragma unroll
      for (int i = 0; i < 4; ++i) { x[i] = pbase[p * DD + c[i]]; a += x[i]; b += x[i] * x[i]; }
      wred2(a, b);
      const float pmu  = a * (1.f / 128.f);
      const float pvar = b * (1.f / 128.f) - pmu * pmu;
      const float pinv = rsqrtf(pvar + LNE);
      float pq = 0.f, dq = 0.f;
      #pragma unroll
      for (int i = 0; i < 4; ++i) {
        const float y = (x[i] - pmu) * pinv * pg[c[i]] + pbv[c[i]];
        pn[p][i] = y;
        pq += y * y;
        dq += q[i] * y;
      }
      wred2(pq, dq);
      const float pscale = 1.f / fmaxf(sqrtf(pq), 1e-6f);
      sim[p] = clip(dq * qscale * pscale, -30.f, 30.f) / temp;
    }
    // softmax over P
    float m = sim[0];
    #pragma unroll
    for (int p = 1; p < PP; ++p) m = fmaxf(m, sim[p]);
    float ex[PP], z = 0.f;
    #pragma unroll
    for (int p = 0; p < PP; ++p) { ex[p] = expf(sim[p] - m); z += ex[p]; }
    const float zinv = 1.f / z;

    // candidate + gate
    float cand[4], rw[4];
    float g = 0.f;
    #pragma unroll
    for (int i = 0; i < 4; ++i) {
      float s = 0.f;
      #pragma unroll
      for (int p = 0; p < PP; ++p) s += ex[p] * zinv * pn[p][i];
      cand[i] = clip(s, -5.f, 5.f);
      rw[i]   = sRaw[rr][c[i]];
      g += gate_w[c[i]]      * clip(rw[i],   -100.f, 100.f);
      g += gate_w[DD + c[i]] * clip(cand[i], -100.f, 100.f);
    }
    g += gate_w[2 * DD + lane]      * clip(sTe[rr][lane],      -100.f, 100.f);
    g += gate_w[2 * DD + 32 + lane] * clip(sTe[rr][lane + 32], -100.f, 100.f);
    g = wred(g) + gate_b[0];
    g = 1.f / (1.f + expf(-g));

    // update, LN, clip, store
    float u[4], t1 = 0.f, t2 = 0.f;
    #pragma unroll
    for (int i = 0; i < 4; ++i) {
      u[i] = (1.f - g) * rw[i] + g * cand[i];
      t1 += u[i]; t2 += u[i] * u[i];
    }
    wred2(t1, t2);
    const float umu  = t1 * (1.f / 128.f);
    const float uvar = t2 * (1.f / 128.f) - umu * umu;
    const float uinv = rsqrtf(uvar + LNE);
    #pragma unroll
    for (int i = 0; i < 4; ++i) {
      float y = (u[i] - umu) * uinv * cg[c[i]] + cbv[c[i]];
      out[(long long)gr * DD + c[i]] = clip(y, -10.f, 10.f);
    }
  }
}

// ---------------------------------------------------------------------------
extern "C" void kernel_launch(void* const* d_in, const int* in_sizes, int n_in,
                              void* d_out, int out_size, void* d_ws, size_t ws_size,
                              hipStream_t stream) {
  (void)n_in; (void)out_size; (void)ws_size;
  const int*   node_ids = (const int*)  d_in[0];
  const float* ef       = (const float*)d_in[1];
  const float* tvals    = (const float*)d_in[2];
  const float* raw_mem  = (const float*)d_in[3];
  const float* protos   = (const float*)d_in[4];
  const float* pg       = (const float*)d_in[5];
  const float* pb       = (const float*)d_in[6];
  const float* time_w   = (const float*)d_in[7];
  const float* time_b   = (const float*)d_in[8];
  const float* edge_w   = (const float*)d_in[9];
  const float* edge_b   = (const float*)d_in[10];
  const float* query_w  = (const float*)d_in[11];
  const float* query_b  = (const float*)d_in[12];
  const float* cg       = (const float*)d_in[13];
  const float* cb       = (const float*)d_in[14];
  const float* gate_w   = (const float*)d_in[15];
  const float* gate_b   = (const float*)d_in[16];
  const float* temp     = (const float*)d_in[17];
  float* out            = (float*)d_out;

  // workspace layout: Wh[128*256] bf16, Wl[128*256] bf16, bcomb[128] f32
  bf16_t* Wh    = (bf16_t*)d_ws;
  bf16_t* Wl    = Wh + 128 * KC;
  float*  bcomb = (float*)(Wl + 128 * KC);

  fold_weights_kernel<<<128, 256, 0, stream>>>(query_w, query_b, edge_w, edge_b,
                                               Wh, Wl, bcomb);

  const int batch   = in_sizes[0];
  const int nblocks = batch / ROWS;
  fused_memory_update<<<nblocks, 256, 0, stream>>>(
      node_ids, ef, tvals, raw_mem, protos, pg, pb, time_w, time_b,
      Wh, Wl, bcomb, cg, cb, gate_w, gate_b, temp, out);
}
